// HET_RelationalAttLayer_32744830665256
// MI455X (gfx1250) — matrix operand order repaired and checked by reading, not compiled, not run on hardware
//
#include <hip/hip_runtime.h>
#include <hip/hip_bf16.h>

// ---------------------------------------------------------------------------
// Relational GAT layer for MI455X (gfx1250, wave32, WMMA).
//   Phase 1: convert X [N,256] and W [R,H,256,32] to bf16 (W transposed to
//            N-major [R,128,256] so B fragments are contiguous in K).
//   Phase 2: per-relation GEMM feat[r] = X @ W_r via v_wmma_f32_16x16x32_bf16.
//            Whole W_r (64KB) staged in LDS via global_load_async_to_lds_b128
//            (ASYNCcnt path), 8 waves/block, 16x128 strip per wave.
//   Phase 3: edge logits + leaky relu + segment-max (atomicMax f32).
//   Phase 4: exp / segment-sum (atomicAdd f32).
//   Phase 5: alpha-weighted scatter-add aggregation (atomicAdd f32).
// ---------------------------------------------------------------------------

typedef __attribute__((ext_vector_type(8)))  float  v8f;
typedef __attribute__((ext_vector_type(16))) __bf16 v16bf;
typedef __attribute__((ext_vector_type(4)))  int    v4i_;
typedef __attribute__((address_space(1))) v4i_ gv4i;   // global AS pointee
typedef __attribute__((address_space(3))) v4i_ sv4i;   // LDS AS pointee

union FragAB { v16bf v; uint4 q[2]; };

#if defined(__has_builtin)
#if __has_builtin(__builtin_amdgcn_global_load_async_to_lds_b128) && \
    __has_builtin(__builtin_amdgcn_s_wait_asynccnt)
#define RGAT_ASYNC_LDS 1
#endif
#endif

__device__ __forceinline__ unsigned short f2bf(float f) {
  unsigned int u = __float_as_uint(f);
  u += 0x7FFFu + ((u >> 16) & 1u);          // round-to-nearest-even
  return (unsigned short)(u >> 16);
}

// ---------------- Phase 1: convert + transpose weights ----------------
__global__ __launch_bounds__(256) void rgat_convert(
    const float* __restrict__ x, const float* __restrict__ cw,
    unsigned short* __restrict__ xb, unsigned short* __restrict__ wbT,
    int N, int R) {
  int i = blockIdx.x * 256 + threadIdx.x;
  if (i < N * 256) xb[i] = f2bf(x[i]);
  if (i < R * 128 * 256) {
    int k = i & 255;            // K index 0..255
    int n = (i >> 8) & 127;     // output column = h*32 + d
    int r = i >> 15;            // relation
    // cw layout [R,H,IN,D]: element (r, n>>5, k, n&31)
    float w = cw[(((size_t)(r * 4 + (n >> 5))) * 256 + k) * 32 + (n & 31)];
    wbT[i] = f2bf(w);           // wbT[(r*128+n)*256 + k]
  }
}

// ---------------- init: out = bias, emax = -inf, denom = 0 ----------------
__global__ __launch_bounds__(256) void rgat_init(
    float* __restrict__ out, const float* __restrict__ bias,
    float* __restrict__ emax, float* __restrict__ denom, int N) {
  int i = blockIdx.x * 256 + threadIdx.x;
  if (i < N * 128) out[i] = bias[i & 127];
  if (i < N * 4) { emax[i] = -__builtin_inff(); denom[i] = 0.0f; }
}

// ---------------- Phase 2: WMMA GEMM, one relation per blockIdx.y ----------
__global__ __launch_bounds__(256) void rgat_gemm(
    const unsigned short* __restrict__ xb,   // [N,256] bf16
    const unsigned short* __restrict__ wbT,  // [R,128,256] bf16 (N-major)
    float* __restrict__ feat,                // [R,N,128] f32
    int N) {
  __shared__ unsigned short ldsW[128 * 256];   // 64 KB: whole W_r^T
  const int r    = blockIdx.y;
  const int tid  = threadIdx.x;
  const int lane = tid & 31;
  const int wave = tid >> 5;

  // Stage W_r^T into LDS (4096 x 16B chunks, 16 per thread).
  {
    const uint4* wsrc = (const uint4*)(wbT + (size_t)r * 128 * 256);
    uint4* wdst = (uint4*)ldsW;
#if RGAT_ASYNC_LDS
    for (int i = tid; i < 4096; i += 256) {
      __builtin_amdgcn_global_load_async_to_lds_b128(
          (gv4i*)(wsrc + i), (sv4i*)(wdst + i), 0, 0);
    }
    __builtin_amdgcn_s_wait_asynccnt(0);
#else
    for (int i = tid; i < 4096; i += 256) wdst[i] = wsrc[i];
#endif
  }
  __syncthreads();

  const int rowBase = (blockIdx.x * 8 + wave) * 16;
  int aRow = rowBase + (lane & 15);
  if (aRow > N - 1) aRow = N - 1;              // clamp tail reads
  const int kHiA = (lane >> 4) * 16;           // A: hi lanes hold K+16 block
  const int nCol = lane & 15;
  const int kHiB = (lane >> 4) * 16;           // B: hi lanes hold K+16 block

  v8f acc[8];
#pragma unroll
  for (int i = 0; i < 8; ++i) acc[i] = (v8f){0.f,0.f,0.f,0.f,0.f,0.f,0.f,0.f};

  const unsigned short* arow = xb + (size_t)aRow * 256;
#pragma unroll
  for (int k0 = 0; k0 < 256; k0 += 32) {
    FragAB fa;
    fa.q[0] = *(const uint4*)(arow + k0 + kHiA);        // K .. K+7
    fa.q[1] = *(const uint4*)(arow + k0 + kHiA + 8);    // K+8 .. K+15
#pragma unroll
    for (int nt = 0; nt < 8; ++nt) {
      FragAB fb;
      const unsigned short* bp = ldsW + ((nt * 16 + nCol) * 256 + k0 + kHiB);
      fb.q[0] = *(const uint4*)(bp);       // K .. K+7
      fb.q[1] = *(const uint4*)(bp + 8);   // K+8 .. K+15
      acc[nt] = __builtin_amdgcn_wmma_f32_16x16x32_bf16(
          false, fa.v, false, fb.v, (short)0, acc[nt], false, false);
    }
  }

  // C/D layout: VGPR p, lanes 0-15 -> row p, lanes 16-31 -> row p+8
  float* fout = feat + (size_t)r * N * 128;
  const int rowOff = (lane >> 4) << 3;
#pragma unroll
  for (int nt = 0; nt < 8; ++nt) {
#pragma unroll
    for (int p = 0; p < 8; ++p) {
      int rw = rowBase + p + rowOff;
      if (rw < N) fout[(size_t)rw * 128 + nt * 16 + nCol] = acc[nt][p];
    }
  }
}

// ---------------- Phase 3: edge logits + segment max ----------------
__global__ __launch_bounds__(256) void rgat_logits(
    const float* __restrict__ feat, const float* __restrict__ attn_l,
    const float* __restrict__ attn_r, const int* __restrict__ src,
    const int* __restrict__ dst, const int* __restrict__ etype,
    float* __restrict__ ebuf, float* __restrict__ emax, int N, int E) {
  int e = blockIdx.x * 8 + (threadIdx.x >> 5);   // one wave per edge
  if (e >= E) return;
  int lane = threadIdx.x & 31;
  int rt = etype[e], s = src[e], d = dst[e];
  const float* fs = feat + ((size_t)rt * N + s) * 128 + lane * 4;
  const float* fd = feat + ((size_t)rt * N + d) * 128 + lane * 4;
  const float* al = attn_l + rt * 128 + lane * 4;   // [R,H,D] flat = lane*4
  const float* ar = attn_r + rt * 128 + lane * 4;
  float4 vs = *(const float4*)fs, vd = *(const float4*)fd;
  float4 va = *(const float4*)al, vb = *(const float4*)ar;
  float p = vs.x*va.x + vs.y*va.y + vs.z*va.z + vs.w*va.w
          + vd.x*vb.x + vd.y*vb.y + vd.z*vb.z + vd.w*vb.w;
  p += __shfl_xor(p, 1);
  p += __shfl_xor(p, 2);
  p += __shfl_xor(p, 4);                 // reduced within 8-lane head group
  if ((lane & 7) == 0) {
    float ev = p > 0.0f ? p : 0.2f * p;  // leaky relu
    int h = lane >> 3;
    ebuf[e * 4 + h] = ev;
    atomicMax(&emax[d * 4 + h], ev);     // global_atomic_max_num_f32
  }
}

// ---------------- Phase 4: exp + segment sum ----------------
__global__ __launch_bounds__(256) void rgat_expsum(
    const int* __restrict__ dst, const float* __restrict__ emax,
    float* __restrict__ ebuf, float* __restrict__ denom, int E) {
  int i = blockIdx.x * 256 + threadIdx.x;
  if (i >= E * 4) return;
  int e = i >> 2, h = i & 3;
  int d = dst[e];
  float m = emax[d * 4 + h];
  if (m < -1e37f) m = 0.0f;              // reference: non-finite -> 0
  float a = __expf(ebuf[i] - m);
  ebuf[i] = a;
  atomicAdd(&denom[d * 4 + h], a);
}

// ---------------- Phase 5: weighted scatter-add aggregation ----------------
__global__ __launch_bounds__(256) void rgat_aggregate(
    const float* __restrict__ feat, const float* __restrict__ ebuf,
    const float* __restrict__ denom, const int* __restrict__ src,
    const int* __restrict__ dst, const int* __restrict__ etype,
    float* __restrict__ out, int N, int E) {
  int e = blockIdx.x * 8 + (threadIdx.x >> 5);   // one wave per edge
  if (e >= E) return;
  int lane = threadIdx.x & 31;
  int rt = etype[e], s = src[e], d = dst[e];
  int h = lane >> 3;
  float alpha = ebuf[e * 4 + h] / denom[d * 4 + h];
  float4 vs = *(const float4*)(feat + ((size_t)rt * N + s) * 128 + lane * 4);
  float* o = out + (size_t)d * 128 + lane * 4;
  atomicAdd(o + 0, vs.x * alpha);
  atomicAdd(o + 1, vs.y * alpha);
  atomicAdd(o + 2, vs.z * alpha);
  atomicAdd(o + 3, vs.w * alpha);
}

// ---------------------------------------------------------------------------
extern "C" void kernel_launch(void* const* d_in, const int* in_sizes, int n_in,
                              void* d_out, int out_size, void* d_ws, size_t ws_size,
                              hipStream_t stream) {
  (void)n_in; (void)out_size; (void)ws_size;
  const float* x      = (const float*)d_in[0];
  const float* cw     = (const float*)d_in[1];
  const float* attn_l = (const float*)d_in[2];
  const float* attn_r = (const float*)d_in[3];
  const float* h_bias = (const float*)d_in[4];
  const int*   src    = (const int*)d_in[5];
  const int*   dst    = (const int*)d_in[6];
  const int*   etype  = (const int*)d_in[7];
  float* out = (float*)d_out;

  const int N = in_sizes[0] / 256;           // 50000
  const int R = in_sizes[1] / (256 * 128);   // 8
  const int E = in_sizes[5];                 // 800000

  // Workspace carve-out (256B aligned slices)
  char* ws = (char*)d_ws;
  size_t off = 0;
  auto take = [&](size_t bytes) {
    void* p = ws + off;
    off += (bytes + 255) & ~(size_t)255;
    return p;
  };
  unsigned short* xb    = (unsigned short*)take((size_t)N * 256 * 2);
  unsigned short* wbT   = (unsigned short*)take((size_t)R * 128 * 256 * 2);
  float*          feat  = (float*)take((size_t)R * N * 128 * 4);
  float*          ebuf  = (float*)take((size_t)E * 4 * 4);
  float*          emax  = (float*)take((size_t)N * 4 * 4);
  float*          denom = (float*)take((size_t)N * 4 * 4);

  int convN = N * 256 > R * 128 * 256 ? N * 256 : R * 128 * 256;
  rgat_convert<<<(convN + 255) / 256, 256, 0, stream>>>(x, cw, xb, wbT, N, R);
  rgat_init<<<(N * 128 + 255) / 256, 256, 0, stream>>>(out, h_bias, emax, denom, N);

  dim3 g((N + 127) / 128, R);
  rgat_gemm<<<g, 256, 0, stream>>>(xb, wbT, feat, N);

  rgat_logits<<<(E + 7) / 8, 256, 0, stream>>>(feat, attn_l, attn_r, src, dst,
                                               etype, ebuf, emax, N, E);
  rgat_expsum<<<(E * 4 + 255) / 256, 256, 0, stream>>>(dst, emax, ebuf, denom, E);
  rgat_aggregate<<<(E + 7) / 8, 256, 0, stream>>>(feat, ebuf, denom, src, dst,
                                                  etype, out, N, E);
}
